// MambaRNN_38457137168404
// MI455X (gfx1250) — compile-verified, matrix operand order checked by reading, not run on hardware
//
#include <hip/hip_runtime.h>
#include <hip/hip_bf16.h>
#include <math.h>

// ---------------------------------------------------------------------------
// Model constants (from the reference)
// ---------------------------------------------------------------------------
constexpr int NL = 24, Dm = 768, Di = 1536, Nst = 16, Rr = 48, Kc = 4;
constexpr int Vv = 50280, NHd = 12, HB = 4, SQ = 256;
constexpr int BT = HB * SQ;          // 1024 backbone tokens
constexpr int MR = HB * NL;          // 96 refine tokens
constexpr int DBL = Rr + 2 * Nst;    // 80
constexpr int HD = Dm / NHd;         // 64

typedef __attribute__((ext_vector_type(16))) _Float16 v16h;
typedef __attribute__((ext_vector_type(8)))  float    v8f;

union HFrag { v16h v; _Float16 h[16]; };
union CFrag { v8f  v; float    f[8];  };
union H2    { _Float16 h[2]; unsigned int u; };
union H4    { _Float16 h[4]; unsigned long long u64; };

// ---------------------------------------------------------------------------
// Generic WMMA GEMM:  Y[M,E] = epi( X[M,K](stride ldx) * W[E,K]^T + bias ) + add
// Block = 128 threads (4 waves). Block tile 64x64, wave tile 16x64 (4 WMMAs).
// Tiles are staged into LDS pre-swizzled into WMMA fragment order so each lane
// fetches its whole 16-half fragment as one contiguous 32B LDS load
// (2x ds_load_b128).
//
// ISA 16-bit fragment layout (A 16x32 / B 32x16, wave32):
//   lane = (m|n)%16 + (k%16 >= 8 ? 16 : 0)
//   half = ((k>>4)<<3) | (k&7)          (even/odd k -> adjacent halves)
//
// Fast path (block-uniform: full interior tile, K%32==0): each thread owns one
// 4-aligned k-quad of one row -> float4 global load + packed ds_store_b64 into
// a loop-invariant LDS slot (frag q at byte offset q*1024). No exec-mask ops.
// act: 0=none, 1=softplus, 2=gelu(exact)
// ---------------------------------------------------------------------------
__global__ __launch_bounds__(128)
void k_gemm_wmma(const float* __restrict__ X, const float* __restrict__ W,
                 float* __restrict__ Y, const float* __restrict__ bias,
                 const float* __restrict__ addsrc,
                 int M, int K, int E, int ldx, int act)
{
    __shared__ unsigned int AswF[1024];  // [frag][lane][half-pair] packed half2
    __shared__ unsigned int BswF[1024];

    const int tid  = threadIdx.x;
    const int wave = tid >> 5;
    const int lane = tid & 31;
    const int m0   = blockIdx.y * 64;
    const int n0   = blockIdx.x * 64;

    v8f acc[4] = {};

    auto mma = [&]() {
        HFrag a;
        a.v = *reinterpret_cast<const v16h*>(&AswF[(wave * 32 + lane) * 8]);
#pragma unroll
        for (int nt = 0; nt < 4; ++nt) {
            HFrag b;
            b.v = *reinterpret_cast<const v16h*>(&BswF[(nt * 32 + lane) * 8]);
            acc[nt] = __builtin_amdgcn_wmma_f32_16x16x32_f16(
                false, a.v, false, b.v, (short)0, acc[nt], false, false);
        }
    };

    const bool full = (m0 + 64 <= M) && (n0 + 64 <= E) && ((K & 31) == 0);
    if (full) {
        // thread -> (row rr within 16, k-quad qc); quad q covers fragment q.
        const int qc = (tid & 7) << 2;          // 0,4,...,28
        const int rr = tid >> 3;                // 0..15
        const int fl = rr | (((qc & 15) >> 3) << 4);
        const unsigned slot = (unsigned)(fl * 8 + ((((qc >> 4) << 3) | (qc & 7)) >> 1));
        const float* aP = X + (size_t)(m0 + rr) * ldx + qc;
        const float* bP = W + (size_t)(n0 + rr) * K + qc;

        for (int k0 = 0; k0 < K; k0 += 32) {
#pragma unroll
            for (int q = 0; q < 4; ++q) {
                float4 av = *reinterpret_cast<const float4*>(aP + (size_t)q * 16 * ldx);
                H4 pa;
                pa.h[0] = (_Float16)av.x; pa.h[1] = (_Float16)av.y;
                pa.h[2] = (_Float16)av.z; pa.h[3] = (_Float16)av.w;
                *reinterpret_cast<unsigned long long*>(&AswF[q * 256 + slot]) = pa.u64;
                float4 bv = *reinterpret_cast<const float4*>(bP + (size_t)q * 16 * K);
                H4 pb;
                pb.h[0] = (_Float16)bv.x; pb.h[1] = (_Float16)bv.y;
                pb.h[2] = (_Float16)bv.z; pb.h[3] = (_Float16)bv.w;
                *reinterpret_cast<unsigned long long*>(&BswF[q * 256 + slot]) = pb.u64;
                if (k0 + 32 < K)                 // keep next K-tile in flight (L2)
                    __builtin_prefetch(bP + (size_t)q * 16 * K + 32, 0, 3);
            }
            aP += 32; bP += 32;
            __syncthreads();
            mma();
            __syncthreads();
        }
    } else {
        for (int k0 = 0; k0 < K; k0 += 32) {
            for (int p = tid; p < 1024; p += 128) {
                int r  = p >> 4;             // row within tile 0..63
                int c  = (p & 15) << 1;      // even k 0..30
                int gm = m0 + r, gk = k0 + c;
                float v0 = (gm < M && gk     < K) ? X[(size_t)gm * ldx + gk]     : 0.f;
                float v1 = (gm < M && gk + 1 < K) ? X[(size_t)gm * ldx + gk + 1] : 0.f;
                H2 pk; pk.h[0] = (_Float16)v0; pk.h[1] = (_Float16)v1;
                int fl = (r & 15) | (((c & 15) >> 3) << 4);
                int hp = ((((c >> 4) << 3) | (c & 7)) >> 1);
                AswF[(r >> 4) * 256 + fl * 8 + hp] = pk.u;
            }
            for (int p = tid; p < 1024; p += 128) {
                int n  = p >> 4;             // col within tile 0..63
                int c  = (p & 15) << 1;
                int ge = n0 + n, gk = k0 + c;
                float v0 = (ge < E && gk     < K) ? W[(size_t)ge * K + gk]     : 0.f;
                float v1 = (ge < E && gk + 1 < K) ? W[(size_t)ge * K + gk + 1] : 0.f;
                H2 pk; pk.h[0] = (_Float16)v0; pk.h[1] = (_Float16)v1;
                int fl = (n & 15) | (((c & 15) >> 3) << 4);
                int hp = ((((c >> 4) << 3) | (c & 7)) >> 1);
                BswF[(n >> 4) * 256 + fl * 8 + hp] = pk.u;
            }
            __syncthreads();
            mma();
            __syncthreads();
        }
    }

    // C/D layout: VGPR r -> M = r (lanes 0-15) or r+8 (lanes 16-31); N = lane%16
#pragma unroll
    for (int nt = 0; nt < 4; ++nt) {
        CFrag cf; cf.v = acc[nt];
        int col   = n0 + (nt << 4) + (lane & 15);
        int rbase = m0 + (wave << 4) + ((lane < 16) ? 0 : 8);
        if (col < E) {
            float bv = bias ? bias[col] : 0.f;
#pragma unroll
            for (int r = 0; r < 8; ++r) {
                int row = rbase + r;
                if (row < M) {
                    float v = cf.f[r] + bv;
                    if (act == 1) v = (v > 20.f) ? v : log1pf(expf(v));
                    else if (act == 2) v = 0.5f * v * (1.f + erff(v * 0.70710678118f));
                    if (addsrc) v += addsrc[(size_t)row * E + col];
                    Y[(size_t)row * E + col] = v;
                }
            }
        }
    }
}

// ---------------------------------------------------------------------------
// Elementwise / reduction kernels
// ---------------------------------------------------------------------------
__device__ __forceinline__ float blockReduceSum(float v, float* sbuf) {
    int tid = threadIdx.x;
    sbuf[tid] = v; __syncthreads();
    for (int s = 128; s > 0; s >>= 1) {
        if (tid < s) sbuf[tid] += sbuf[tid + s];
        __syncthreads();
    }
    float r = sbuf[0]; __syncthreads();
    return r;
}

__global__ void k_embed(const int* __restrict__ ids, const int* __restrict__ mask,
                        const float* __restrict__ embed, float* __restrict__ h) {
    int i = blockIdx.x * blockDim.x + threadIdx.x;
    if (i >= BT * Dm) return;
    int t = i / Dm, d = i % Dm;
    h[i] = embed[(size_t)ids[t] * Dm + d] * (float)mask[t];
}

__global__ __launch_bounds__(256)
void k_rmsnorm(const float* __restrict__ x, const float* __restrict__ w,
               float* __restrict__ y, int cols) {
    __shared__ float sbuf[256];
    const float* xr = x + (size_t)blockIdx.x * cols;
    float s = 0.f;
    for (int c = threadIdx.x; c < cols; c += 256) { float v = xr[c]; s += v * v; }
    s = blockReduceSum(s, sbuf);
    float inv = rsqrtf(s / cols + 1e-5f);
    for (int c = threadIdx.x; c < cols; c += 256)
        y[(size_t)blockIdx.x * cols + c] = xr[c] * inv * w[c];
}

__global__ __launch_bounds__(256)
void k_layernorm(const float* __restrict__ x, const float* __restrict__ w,
                 const float* __restrict__ b, float* __restrict__ y, int cols) {
    __shared__ float sbuf[256];
    const float* xr = x + (size_t)blockIdx.x * cols;
    float s = 0.f;
    for (int c = threadIdx.x; c < cols; c += 256) s += xr[c];
    s = blockReduceSum(s, sbuf);
    float mean = s / cols;
    float s2 = 0.f;
    for (int c = threadIdx.x; c < cols; c += 256) { float v = xr[c] - mean; s2 += v * v; }
    s2 = blockReduceSum(s2, sbuf);
    float inv = rsqrtf(s2 / cols + 1e-5f);
    for (int c = threadIdx.x; c < cols; c += 256)
        y[(size_t)blockIdx.x * cols + c] = (xr[c] - mean) * inv * w[c] + b[c];
}

// causal depthwise conv (K=4) + bias + SiLU.  x is the first half of xz rows.
__global__ void k_conv_silu(const float* __restrict__ xz, const float* __restrict__ cw,
                            const float* __restrict__ cb, float* __restrict__ xc) {
    int i = blockIdx.x * blockDim.x + threadIdx.x;
    if (i >= BT * Di) return;
    int t = i / Di, d = i % Di;
    int b = t / SQ, l = t % SQ;
    float s = cb[d];
#pragma unroll
    for (int k = 0; k < Kc; ++k) {
        int l2 = l - (Kc - 1) + k;
        if (l2 >= 0)
            s += xz[((size_t)(b * SQ + l2)) * (2 * Di) + d] * cw[d * Kc + k];
    }
    xc[i] = s / (1.f + expf(-s));   // silu
}

// Selective scan: one thread per (batch, channel), N=16 state in registers.
__global__ __launch_bounds__(256)
void k_scan(const float* __restrict__ delta, const float* __restrict__ xc,
            const float* __restrict__ dbl, const float* __restrict__ A_log,
            const float* __restrict__ init, float* __restrict__ y,
            float* __restrict__ last) {
    int i = blockIdx.x * blockDim.x + threadIdx.x;
    if (i >= HB * Di) return;
    int b = i / Di, d = i % Di;
    float hs[Nst], Av[Nst];
#pragma unroll
    for (int n = 0; n < Nst; ++n) {
        hs[n] = init ? init[((size_t)(b * Di + d)) * Nst + n] : 0.f;
        Av[n] = -expf(A_log[d * Nst + n]);
    }
    for (int l = 0; l < SQ; ++l) {
        size_t t = (size_t)b * SQ + l;
        float dt = delta[t * Di + d];
        float xv = xc[t * Di + d];
        const float* Bm = dbl + t * DBL + Rr;
        const float* Cm = dbl + t * DBL + Rr + Nst;
        float acc = 0.f;
        float dx = dt * xv;
#pragma unroll
        for (int n = 0; n < Nst; ++n) {
            hs[n] = expf(dt * Av[n]) * hs[n] + dx * Bm[n];
            acc += hs[n] * Cm[n];
        }
        y[t * Di + d] = acc;
    }
#pragma unroll
    for (int n = 0; n < Nst; ++n) last[((size_t)(b * Di + d)) * Nst + n] = hs[n];
}

// y = (scan_y + xc * D_p) * silu(z)   (z = second half of xz rows)
__global__ void k_gate(const float* __restrict__ ys, const float* __restrict__ xc,
                       const float* __restrict__ xz, const float* __restrict__ dp,
                       float* __restrict__ yg) {
    int i = blockIdx.x * blockDim.x + threadIdx.x;
    if (i >= BT * Di) return;
    int t = i / Di, d = i % Di;
    float z = xz[(size_t)t * (2 * Di) + Di + d];
    float sz = z / (1.f + expf(-z));
    yg[i] = (ys[i] + xc[i] * dp[d]) * sz;
}

// Attention over the 24-layer "sequence" (tiny): one block per (b, head).
__global__ __launch_bounds__(32)
void k_attn(const float* __restrict__ qkv, float* __restrict__ o) {
    int bh = blockIdx.x;
    int b = bh / NHd, hh = bh % NHd;
    __shared__ float Ks[NL][HD], Vs[NL][HD];
    for (int i = threadIdx.x; i < NL * HD; i += 32) {
        int s = i / HD, dd = i % HD;
        size_t row = (size_t)(b * NL + s) * (3 * Dm);
        Ks[s][dd] = qkv[row + Dm + hh * HD + dd];
        Vs[s][dd] = qkv[row + 2 * Dm + hh * HD + dd];
    }
    __syncthreads();
    int q = threadIdx.x;
    if (q < NL) {
        float qv[HD];
        size_t row = (size_t)(b * NL + q) * (3 * Dm);
        for (int dd = 0; dd < HD; ++dd) qv[dd] = qkv[row + hh * HD + dd];
        float sc[NL], mx = -1e30f;
        for (int s = 0; s < NL; ++s) {
            float dot = 0.f;
            for (int dd = 0; dd < HD; ++dd) dot += qv[dd] * Ks[s][dd];
            sc[s] = dot * 0.125f;               // 1/sqrt(64)
            mx = fmaxf(mx, sc[s]);
        }
        float sum = 0.f;
        for (int s = 0; s < NL; ++s) { sc[s] = expf(sc[s] - mx); sum += sc[s]; }
        float inv = 1.f / sum;
        for (int dd = 0; dd < HD; ++dd) {
            float a = 0.f;
            for (int s = 0; s < NL; ++s) a += sc[s] * Vs[s][dd];
            o[(size_t)(b * NL + q) * Dm + hh * HD + dd] = a * inv;
        }
    }
}

// d1[b,s,di] = sum_n states[s][b][di][n] * C_w[n]
__global__ void k_sreduce(const float* __restrict__ states, const float* __restrict__ Cw,
                          float* __restrict__ d1) {
    int i = blockIdx.x * blockDim.x + threadIdx.x;
    if (i >= HB * NL * Di) return;
    int b = i / (NL * Di), s = (i / Di) % NL, di = i % Di;
    const float* st = states + ((size_t)(s * HB + b) * Di + di) * Nst;
    float a = 0.f;
#pragma unroll
    for (int n = 0; n < Nst; ++n) a += st[n] * Cw[n];
    d1[(size_t)(b * NL + s) * Di + di] = a;
}

// states[s][b][di][n] += dback[b,s,di] * B_w[n]
__global__ void k_sadd(float* __restrict__ states, const float* __restrict__ dback,
                       const float* __restrict__ Bw) {
    int i = blockIdx.x * blockDim.x + threadIdx.x;
    if (i >= HB * NL * Di) return;
    int b = i / (NL * Di), s = (i / Di) % NL, di = i % Di;
    float dv = dback[(size_t)(b * NL + s) * Di + di];
    float* st = states + ((size_t)(s * HB + b) * Di + di) * Nst;
#pragma unroll
    for (int n = 0; n < Nst; ++n) st[n] += dv * Bw[n];
}

// ---------------------------------------------------------------------------
// Host orchestration
// ---------------------------------------------------------------------------
static inline void gemm(hipStream_t st, const float* X, int ldx, const float* W,
                        float* Y, const float* bias, const float* addsrc,
                        int M, int K, int E, int act) {
    dim3 grid((E + 63) / 64, (M + 63) / 64);
    k_gemm_wmma<<<grid, 128, 0, st>>>(X, W, Y, bias, addsrc, M, K, E, ldx, act);
}

extern "C" void kernel_launch(void* const* d_in, const int* in_sizes, int n_in,
                              void* d_out, int out_size, void* d_ws, size_t ws_size,
                              hipStream_t stream) {
    const int*   q_ids  = (const int*)d_in[0];
    const int*   q_mask = (const int*)d_in[1];
    const int*   a_ids  = (const int*)d_in[2];
    const int*   a_mask = (const int*)d_in[3];
    const float* embed  = (const float*)d_in[4];
    const float* norm_w = (const float*)d_in[5];
    const float* ipw    = (const float*)d_in[6];
    const float* cw     = (const float*)d_in[7];
    const float* cb     = (const float*)d_in[8];
    const float* xpw    = (const float*)d_in[9];
    const float* dpw    = (const float*)d_in[10];
    const float* dpb    = (const float*)d_in[11];
    const float* alog   = (const float*)d_in[12];
    const float* Dp     = (const float*)d_in[13];
    const float* opw    = (const float*)d_in[14];
    const float* norm_f = (const float*)d_in[15];
    const float* ln1w   = (const float*)d_in[16];
    const float* ln1b   = (const float*)d_in[17];
    const float* qkvw   = (const float*)d_in[18];
    const float* qkvb_w = (const float*)d_in[19];
    const float* aow    = (const float*)d_in[20];
    const float* aob    = (const float*)d_in[21];
    const float* ln2w   = (const float*)d_in[22];
    const float* ln2b   = (const float*)d_in[23];
    const float* ff1w   = (const float*)d_in[24];
    const float* ff1b   = (const float*)d_in[25];
    const float* ff2w   = (const float*)d_in[26];
    const float* ff2b   = (const float*)d_in[27];
    const float* Cw     = (const float*)d_in[28];
    const float* pow_   = (const float*)d_in[29];
    const float* Bw     = (const float*)d_in[30];
    const float* piw    = (const float*)d_in[31];
    float* out = (float*)d_out;

    // bump-allocate workspace (floats)
    float* ws = (float*)d_ws;
    size_t off = 0;
    auto alloc = [&](size_t n) { float* p = ws + off; off += n; return p; };
    float* h      = alloc((size_t)BT * Dm);
    float* xn     = alloc((size_t)BT * Dm);
    float* xz     = alloc((size_t)BT * 2 * Di);
    float* xc     = alloc((size_t)BT * Di);
    float* dbl    = alloc((size_t)BT * DBL);
    float* delta  = alloc((size_t)BT * Di);
    float* ys     = alloc((size_t)BT * Di);
    float* yg     = alloc((size_t)BT * Di);
    float* states = alloc((size_t)NL * HB * Di * Nst);
    float* hn     = alloc((size_t)BT * Dm);
    float* d1     = alloc((size_t)MR * Di);
    float* dproj  = alloc((size_t)MR * Dm);
    float* ybuf   = alloc((size_t)MR * Dm);
    float* qkvb   = alloc((size_t)MR * 3 * Dm);
    float* obuf   = alloc((size_t)MR * Dm);
    float* x2     = alloc((size_t)MR * Dm);
    float* ffb    = alloc((size_t)MR * 4 * Dm);
    float* tfo    = alloc((size_t)MR * Dm);
    float* dback  = alloc((size_t)MR * Di);
    (void)ws_size; (void)in_sizes; (void)n_in; (void)out_size;

    auto run_backbone = [&](const int* ids, const int* mask, bool use_init,
                            bool emit_logits) {
        k_embed<<<(BT * Dm + 255) / 256, 256, 0, stream>>>(ids, mask, embed, h);
        for (int l = 0; l < NL; ++l) {
            const float* l_ipw  = ipw + (size_t)l * 2 * Di * Dm;
            const float* l_cw   = cw  + (size_t)l * Di * Kc;
            const float* l_cb   = cb  + (size_t)l * Di;
            const float* l_xpw  = xpw + (size_t)l * DBL * Di;
            const float* l_dpw  = dpw + (size_t)l * Di * Rr;
            const float* l_dpb  = dpb + (size_t)l * Di;
            const float* l_alog = alog + (size_t)l * Di * Nst;
            const float* l_dp   = Dp  + (size_t)l * Di;
            const float* l_opw  = opw + (size_t)l * Dm * Di;
            float* l_state = states + (size_t)l * HB * Di * Nst;

            k_rmsnorm<<<BT, 256, 0, stream>>>(h, norm_w + (size_t)l * Dm, xn, Dm);
            gemm(stream, xn, Dm, l_ipw, xz, nullptr, nullptr, BT, Dm, 2 * Di, 0);
            k_conv_silu<<<(BT * Di + 255) / 256, 256, 0, stream>>>(xz, l_cw, l_cb, xc);
            gemm(stream, xc, Di, l_xpw, dbl, nullptr, nullptr, BT, Di, DBL, 0);
            gemm(stream, dbl, DBL, l_dpw, delta, l_dpb, nullptr, BT, Rr, Di, 1 /*softplus*/);
            k_scan<<<(HB * Di + 255) / 256, 256, 0, stream>>>(
                delta, xc, dbl, l_alog, use_init ? l_state : nullptr, ys, l_state);
            k_gate<<<(BT * Di + 255) / 256, 256, 0, stream>>>(ys, xc, xz, l_dp, yg);
            gemm(stream, yg, Di, l_opw, h, nullptr, h /*residual in-place*/, BT, Di, Dm, 0);
        }
        if (emit_logits) {
            k_rmsnorm<<<BT, 256, 0, stream>>>(h, norm_f, hn, Dm);
            gemm(stream, hn, Dm, embed, out, nullptr, nullptr, BT, Dm, Vv, 0);
        }
    };

    // pass 1: question, zero-init states, collect last states
    run_backbone(q_ids, q_mask, /*use_init=*/false, /*emit_logits=*/false);

    // refinement: 12 iterations of state -> tf-layer -> state update
    for (int it = 0; it < 12; ++it) {
        k_sreduce<<<(HB * NL * Di + 255) / 256, 256, 0, stream>>>(states, Cw, d1);
        gemm(stream, d1, Di, pow_, dproj, nullptr, nullptr, MR, Di, Dm, 0);
        k_layernorm<<<MR, 256, 0, stream>>>(dproj, ln1w, ln1b, ybuf, Dm);
        gemm(stream, ybuf, Dm, qkvw, qkvb, qkvb_w, nullptr, MR, Dm, 3 * Dm, 0);
        k_attn<<<HB * NHd, 32, 0, stream>>>(qkvb, obuf);
        gemm(stream, obuf, Dm, aow, x2, aob, dproj, MR, Dm, Dm, 0);
        k_layernorm<<<MR, 256, 0, stream>>>(x2, ln2w, ln2b, ybuf, Dm);
        gemm(stream, ybuf, Dm, ff1w, ffb, ff1b, nullptr, MR, Dm, 4 * Dm, 2 /*gelu*/);
        gemm(stream, ffb, 4 * Dm, ff2w, tfo, ff2b, x2, MR, 4 * Dm, Dm, 0);
        gemm(stream, tfo, Dm, piw, dback, nullptr, nullptr, MR, Dm, Di, 0);
        k_sadd<<<(HB * NL * Di + 255) / 256, 256, 0, stream>>>(states, dback, Bw);
    }

    // pass 2: answer, refined init states, emit logits into d_out
    run_backbone(a_ids, a_mask, /*use_init=*/true, /*emit_logits=*/true);
}
